// MultiLatentAttention_14937896255595
// MI455X (gfx1250) — matrix operand order repaired and checked by reading, not compiled, not run on hardware
//
#include <hip/hip_runtime.h>
#include <hip/hip_bf16.h>

// ---------------------------------------------------------------------------
// MultiLatentAttention, restructured for MI455X (gfx1250, wave32, WMMA):
//   1. Only t<128 of K/V is ever used (mask t<=l, l<64, chunk=128).
//   2. out[b] = softmax_head(x[b]@Wg * s) @ M[b],  M[b] = attn[b] @ Wp (per head).
// GEMMs use v_wmma_f32_16x16x32_bf16. All B operands are pre-swizzled into the
// contiguous-k (k = 16*half + s) wave32 B-fragment layout so the hot loop is b128 loads + WMMA only.
// Fragment layout (32x16 K-major tile, 256 dwords):
//   dword[(k0/32)*numCT + ct][lane][v],  lane = half*16 + n,
//   dword v = pack(B[k0+half*16+2v][ct*16+n], B[k0+half*16+2v+1][ct*16+n])
// ---------------------------------------------------------------------------

typedef __attribute__((ext_vector_type(16))) __bf16 v16bf;
typedef __attribute__((ext_vector_type(8)))  float  v8f;

#define B_   4
#define T_   4096
#define C_   2048
#define H_   16
#define D_   128
#define L_   64
#define HL_  1024
#define SCALE_ 0.08838834764831845f

union AB { v16bf v; unsigned u[8]; uint4 q[2]; };

static __device__ __forceinline__ unsigned short f2bf(float f) {
  union { float f; unsigned u; } v; v.f = f;
  unsigned r = v.u + 0x7FFFu + ((v.u >> 16) & 1u);
  return (unsigned short)(r >> 16);
}
static __device__ __forceinline__ unsigned pk2(float a, float b) {
  return (unsigned)f2bf(a) | ((unsigned)f2bf(b) << 16);
}
static __device__ __forceinline__ v8f wmma_bf16(const AB& a, const AB& b, v8f c) {
  return __builtin_amdgcn_wmma_f32_16x16x32_bf16(false, a.v, false, b.v,
                                                 (short)0, c, false, false);
}

// -------------------------- fp32 -> bf16 convert ---------------------------
__global__ void cvt_f32_bf16(const float* __restrict__ in,
                             unsigned short* __restrict__ out, int n) {
  int i = blockIdx.x * blockDim.x + threadIdx.x;
  if (i < n) out[i] = f2bf(in[i]);
}

// -------- swizzle a row-major f32 [K x N] weight into B-fragment bf16 -------
__global__ void wswizzle(const float* __restrict__ W, unsigned* __restrict__ out,
                         int numCT, int ldn, int nDwords) {
  int idx = blockIdx.x * blockDim.x + threadIdx.x;
  if (idx >= nDwords) return;
  int v = idx & 7;
  int lane = (idx >> 3) & 31;
  int rest = idx >> 8;
  int ct = rest % numCT;
  int k0f = rest / numCT;
  int n = lane & 15, half = lane >> 4;
  int kk = k0f * 32 + half * 16 + 2 * v;
  int col = ct * 16 + n;
  out[idx] = pk2(W[(size_t)kk * ldn + col], W[(size_t)(kk + 1) * ldn + col]);
}

// --------- K/V GEMM (first 128 rows only): O[b,t,c] = xb[b,t,:] @ W --------
// grid = B*8 (16-row tiles), 256 threads (8 waves); wave: 2 strips of 8 tiles
__global__ void kv_gemm(const unsigned short* __restrict__ xb,  // [B*T, C] bf16
                        const unsigned* __restrict__ Wsw,       // swizzled [64][128][256]
                        float* __restrict__ out) {              // [B,128,C] f32
  int b = blockIdx.x >> 3, rowTile = blockIdx.x & 7;
  int lane = threadIdx.x & 31, wave = threadIdx.x >> 5;
  int m = lane & 15, half = lane >> 4;
  const unsigned short* A0 = xb + ((size_t)b * T_ + rowTile * 16 + m) * C_;
  float* O = out + (size_t)b * 128 * C_;
  for (int strip = wave; strip < 16; strip += 8) {
    int ct0 = strip * 8;
    v8f acc[8] = {};
    for (int k0 = 0; k0 < C_; k0 += 32) {
      AB a;
      const uint4* Ap = (const uint4*)(A0 + k0 + half * 16);
      a.q[0] = Ap[0]; a.q[1] = Ap[1];
      const unsigned* Bbase = Wsw + ((size_t)(k0 >> 5) * 128 + ct0) * 256 + lane * 8;
#pragma unroll
      for (int j = 0; j < 8; ++j) {
        AB bm;
        const uint4* Bp = (const uint4*)(Bbase + j * 256);
        bm.q[0] = Bp[0]; bm.q[1] = Bp[1];
        acc[j] = wmma_bf16(a, bm, acc[j]);
      }
    }
#pragma unroll
    for (int j = 0; j < 8; ++j)
#pragma unroll
      for (int r = 0; r < 8; ++r)
        O[(size_t)(rowTile * 16 + r + half * 8) * C_ + (ct0 + j) * 16 + m] = acc[j][r];
  }
}

// ------------------------------ RoPE in place ------------------------------
__global__ void rope_kernel(float* __restrict__ kf) {  // [B,128,H,D]
  int i = blockIdx.x * blockDim.x + threadIdx.x;
  if (i >= B_ * 128 * H_ * (D_ / 2)) return;
  int d2 = i & 63, h = (i >> 6) & 15, t = (i >> 10) & 127, b = i >> 17;
  float inv = powf(10000.0f, -(float)d2 / 64.0f);
  float ang = (float)t * inv;
  float c = cosf(ang), s = sinf(ang);
  size_t base = ((size_t)b * 128 + t) * C_ + h * D_ + 2 * d2;
  float x0 = kf[base], x1 = kf[base + 1];
  kf[base]     = x0 * c - x1 * s;
  kf[base + 1] = x0 * s + x1 * c;
}

// ------------- attention: masked softmax over first chunk only -------------
__global__ void attn_kernel(const float* __restrict__ lq,        // [L,H,D]
                            const float* __restrict__ kf,        // [B,128,C] roped
                            const float* __restrict__ vf,        // [B,128,C]
                            unsigned short* __restrict__ attn) { // [B,H,L,D] bf16
  __shared__ float sw[L_ * 128];  // 32 KB
  int b = blockIdx.x >> 4, h = blockIdx.x & 15;
  int tid = threadIdx.x;
  const float* K = kf + (size_t)b * 128 * C_ + h * D_;
  const float* V = vf + (size_t)b * 128 * C_ + h * D_;
  for (int idx = tid; idx < L_ * 128; idx += 256) {
    int l = idx >> 7, t = idx & 127;
    float s = -1e9f;
    if (t <= l) {
      const float* q = lq + (size_t)l * (H_ * D_) + h * D_;
      const float* kr = K + (size_t)t * C_;
      float acc = 0.f;
      for (int d = 0; d < D_; ++d) acc += q[d] * kr[d];
      s = acc * SCALE_;
    }
    sw[idx] = s;
  }
  __syncthreads();
  if (tid < L_) {
    float mx = -1e30f;
    for (int t = 0; t < 128; ++t) mx = fmaxf(mx, sw[tid * 128 + t]);
    float sum = 0.f;
    for (int t = 0; t < 128; ++t) {
      float e = expf(sw[tid * 128 + t] - mx);
      sw[tid * 128 + t] = e; sum += e;
    }
    float inv = 1.f / sum;
    for (int t = 0; t < 128; ++t) sw[tid * 128 + t] *= inv;
  }
  __syncthreads();
  for (int idx = tid; idx < L_ * D_; idx += 256) {
    int l = idx >> 7, d = idx & 127;
    float acc = 0.f;
    for (int t = 0; t < 128; ++t) acc += sw[l * 128 + t] * V[(size_t)t * C_ + d];
    attn[((size_t)b * H_ + h) * L_ * D_ + idx] = f2bf(acc);
  }
}

// ------ M build: M[b, h*64+l, :] = attn @ Wp_h, OUTPUT IN FRAGMENT LAYOUT ---
// grid = B*H, 256 threads (8 waves). C-layout -> B-fragment store: lane
// (half_c,n) owns dwords half_c*4..+3 of fragment-lane half_b*16+n (1 b128 st).
__global__ void mbuild_gemm(const unsigned short* __restrict__ attn, // [B,H,L,D]
                            const unsigned* __restrict__ WpSw,       // [64][128][256]
                            unsigned* __restrict__ MbSw) {           // [B][32][128][256]
  int b = blockIdx.x >> 4, h = blockIdx.x & 15;
  int lane = threadIdx.x & 31, wave = threadIdx.x >> 5;
  int m = lane & 15, half = lane >> 4;
  const unsigned short* A0 = attn + ((size_t)b * H_ + h) * L_ * D_;
  for (int u = wave; u < 64; u += 8) {
    int rt = u >> 4, ct0 = (u & 15) * 8;
    const unsigned short* Arow = A0 + (size_t)(rt * 16 + m) * D_;
    v8f acc[8] = {};
    for (int k0 = 0; k0 < D_; k0 += 32) {
      AB a;
      const uint4* Ap = (const uint4*)(Arow + k0 + half * 16);
      a.q[0] = Ap[0]; a.q[1] = Ap[1];
      const unsigned* Bbase = WpSw + ((size_t)(h * 4 + (k0 >> 5)) * 128 + ct0) * 256 + lane * 8;
#pragma unroll
      for (int j = 0; j < 8; ++j) {
        AB bm;
        const uint4* Bp = (const uint4*)(Bbase + j * 256);
        bm.q[0] = Bp[0]; bm.q[1] = Bp[1];
        acc[j] = wmma_bf16(a, bm, acc[j]);
      }
    }
    int fM = h * 2 + (rt >> 1);   // global M-row base (h*64+rt*16) / 32
    int halfB = rt & 1;
#pragma unroll
    for (int j = 0; j < 8; ++j) {
      uint4 st;
      st.x = pk2(acc[j][0], acc[j][1]);
      st.y = pk2(acc[j][2], acc[j][3]);
      st.z = pk2(acc[j][4], acc[j][5]);
      st.w = pk2(acc[j][6], acc[j][7]);
      size_t off = (((size_t)b * 32 + fM) * 128 + (ct0 + j)) * 256
                 + (halfB * 16 + m) * 8 + half * 4;
      *(uint4*)(MbSw + off) = st;
    }
  }
}

// ----- fused tail: out rows = softmax_head(xb @ Wg * s) @ M[b], 16 rows -----
// grid = (B*T)/16 blocks, 256 threads (8 waves). Gates live only in LDS.
__global__ void fused_out(const unsigned short* __restrict__ xb,  // [B*T, C]
                          const unsigned* __restrict__ WgSw,      // [64][64][256]
                          const unsigned* __restrict__ MbSw,      // [B][32][128][256]
                          float* __restrict__ out) {              // [B*T, C]
  __shared__ float G[16 * HL_];  // 64 KB: gate logits f32, then packed bf16
  int rowBase = blockIdx.x * 16;
  int b = rowBase >> 12;  // 4096 rows per batch; blocks never straddle batches
  int lane = threadIdx.x & 31, wave = threadIdx.x >> 5;
  int m = lane & 15, half = lane >> 4;
  const unsigned short* A0 = xb + (size_t)(rowBase + m) * C_;

  // ---- GEMM1: G[16 x 1024] = xb_rows @ Wg ; wave owns 8-col-tile strip ----
  {
    int ct0 = wave * 8;
    v8f acc[8] = {};
    for (int k0 = 0; k0 < C_; k0 += 32) {
      AB a;
      const uint4* Ap = (const uint4*)(A0 + k0 + half * 16);
      a.q[0] = Ap[0]; a.q[1] = Ap[1];
      const unsigned* Bbase = WgSw + ((size_t)(k0 >> 5) * 64 + ct0) * 256 + lane * 8;
#pragma unroll
      for (int j = 0; j < 8; ++j) {
        AB bm;
        const uint4* Bp = (const uint4*)(Bbase + j * 256);
        bm.q[0] = Bp[0]; bm.q[1] = Bp[1];
        acc[j] = wmma_bf16(a, bm, acc[j]);
      }
    }
#pragma unroll
    for (int j = 0; j < 8; ++j)
#pragma unroll
      for (int r = 0; r < 8; ++r)
        G[(r + half * 8) * HL_ + (ct0 + j) * 16 + m] = acc[j][r];
  }
  __syncthreads();

  // ---- per-(row,head) softmax, repacked in place to bf16 (own region) ----
  {
    int r = threadIdx.x >> 4, h = threadIdx.x & 15;
    float* g = &G[r * HL_ + h * L_];
    float mx = -1e30f;
    for (int l = 0; l < L_; ++l) mx = fmaxf(mx, g[l] * SCALE_);
    float sum = 0.f;
    for (int l = 0; l < L_; ++l) {
      float e = expf(g[l] * SCALE_ - mx);
      g[l] = e; sum += e;
    }
    float inv = 1.f / sum;
    unsigned* gdw = (unsigned*)g;
    for (int j = 0; j < L_ / 2; ++j) {  // read 2j,2j+1 then write j: safe
      gdw[j] = pk2(g[2 * j] * inv, g[2 * j + 1] * inv);
    }
  }
  __syncthreads();

  // ---- GEMM2: out[16 x 2048] = gates_bf16(LDS) @ Mb[b] ----
  const unsigned* Gdw = (const unsigned*)G;
  const unsigned* Bb = MbSw + (size_t)b * 32 * 128 * 256;
  for (int strip = wave; strip < 16; strip += 8) {
    int ct0 = strip * 8;
    v8f acc[8] = {};
    for (int k0 = 0; k0 < HL_; k0 += 32) {
      AB a;
      int kk0 = k0 + half * 16;  // never crosses a 64-group within +15
      const uint4* Ap = (const uint4*)&Gdw[m * HL_ + (kk0 >> 6) * 64 + ((kk0 & 63) >> 1)];
      a.q[0] = Ap[0]; a.q[1] = Ap[1];
      const unsigned* Bbase = Bb + ((size_t)(k0 >> 5) * 128 + ct0) * 256 + lane * 8;
#pragma unroll
      for (int j = 0; j < 8; ++j) {
        AB bm;
        const uint4* Bp = (const uint4*)(Bbase + j * 256);
        bm.q[0] = Bp[0]; bm.q[1] = Bp[1];
        acc[j] = wmma_bf16(a, bm, acc[j]);
      }
    }
#pragma unroll
    for (int j = 0; j < 8; ++j)
#pragma unroll
      for (int r = 0; r < 8; ++r)
        out[(size_t)(rowBase + r + half * 8) * C_ + (ct0 + j) * 16 + m] = acc[j][r];
  }
}

// ---------------------------------------------------------------------------
extern "C" void kernel_launch(void* const* d_in, const int* in_sizes, int n_in,
                              void* d_out, int out_size, void* d_ws, size_t ws_size,
                              hipStream_t stream) {
  const float* x  = (const float*)d_in[0];
  const float* lq = (const float*)d_in[1];
  const float* Wk = (const float*)d_in[2];
  const float* Wv = (const float*)d_in[3];
  const float* Wg = (const float*)d_in[4];
  const float* Wp = (const float*)d_in[5];
  float* out = (float*)d_out;
  char* ws = (char*)d_ws;

  size_t off = 0;
  unsigned short* xb = (unsigned short*)(ws + off); off += (size_t)B_ * T_ * C_ * 2;      // 64 MB
  unsigned* WkSw = (unsigned*)(ws + off); off += (size_t)C_ * C_ * 2;                     // 8 MB
  unsigned* WvSw = (unsigned*)(ws + off); off += (size_t)C_ * C_ * 2;                     // 8 MB
  unsigned* WgSw = (unsigned*)(ws + off); off += (size_t)C_ * HL_ * 2;                    // 4 MB
  unsigned* WpSw = (unsigned*)(ws + off); off += (size_t)C_ * C_ * 2;                     // 8 MB
  float* kf      = (float*)(ws + off);    off += (size_t)B_ * 128 * C_ * 4;               // 4 MB
  float* vf      = (float*)(ws + off);    off += (size_t)B_ * 128 * C_ * 4;               // 4 MB
  unsigned short* atb = (unsigned short*)(ws + off); off += (size_t)B_ * H_ * L_ * D_ * 2;// 1 MB
  unsigned* MbSw = (unsigned*)(ws + off); off += (size_t)B_ * HL_ * C_ * 2;               // 16 MB

  // x -> bf16 (A operand; pairs along K are contiguous -> b128 loads)
  { int n = B_ * T_ * C_; cvt_f32_bf16<<<(n + 255) / 256, 256, 0, stream>>>(x, xb, n); }

  // weights -> swizzled B-fragment layout (one pass each, fp32 source)
  { int n = C_ * C_ / 2;  wswizzle<<<(n + 255) / 256, 256, 0, stream>>>(Wk, WkSw, 128, C_, n); }
  { int n = C_ * C_ / 2;  wswizzle<<<(n + 255) / 256, 256, 0, stream>>>(Wv, WvSw, 128, C_, n); }
  { int n = C_ * HL_ / 2; wswizzle<<<(n + 255) / 256, 256, 0, stream>>>(Wg, WgSw, 64, HL_, n); }
  { int n = C_ * C_ / 2;  wswizzle<<<(n + 255) / 256, 256, 0, stream>>>(Wp, WpSw, 128, C_, n); }

  // K,V for the only live chunk (t < 128), then RoPE on K
  kv_gemm<<<B_ * 8, 256, 0, stream>>>(xb, WkSw, kf);
  kv_gemm<<<B_ * 8, 256, 0, stream>>>(xb, WvSw, vf);
  rope_kernel<<<(B_ * 128 * H_ * (D_ / 2)) / 256, 256, 0, stream>>>(kf);

  // masked-softmax attention -> attn bf16
  attn_kernel<<<B_ * H_, 256, 0, stream>>>(lq, kf, vf, atb);

  // M[b] = attn @ Wp (block-diagonal over heads), written in fragment layout
  mbuild_gemm<<<B_ * H_, 256, 0, stream>>>(atb, WpSw, MbSw);

  // out = softmax_head(x @ Wg * s) @ M[b], gates fused in LDS
  fused_out<<<(B_ * T_) / 16, 256, 0, stream>>>(xb, WgSw, MbSw, out);
}
